// RNNAttentionDecoder_42880953484347
// MI455X (gfx1250) — matrix operand order repaired and checked by reading, not compiled, not run on hardware
//
#include <hip/hip_runtime.h>
#include <hip/hip_bf16.h>

// Problem sizes (fixed by the reference).
#define V_ 32000
#define H_ 512
#define B_ 32
#define T_ 64
#define S_ 64

typedef __attribute__((ext_vector_type(16))) __bf16 v16bf;
typedef __attribute__((ext_vector_type(8)))  float  v8f;
typedef __attribute__((ext_vector_type(4)))  unsigned int v4u;
typedef __attribute__((ext_vector_type(8)))  int    v8i;
typedef __attribute__((ext_vector_type(4)))  int    v4i;

// ---------------------------------------------------------------------------
// helpers
// ---------------------------------------------------------------------------
__device__ __forceinline__ unsigned short f2bf_us(float f) {
  union { float f; unsigned u; } v; v.f = f;
  unsigned r = v.u + 0x7FFFu + ((v.u >> 16) & 1u);   // round-to-nearest-even
  return (unsigned short)(r >> 16);
}

// A-matrix fragment (16x32 bf16, dense layout per CDNA5 ISA 7.12.2)
// lane<16: row m=lane, K = {k0..k0+7} U {k0+16..k0+23}
// lane>=16: row m=lane-16, K = {k0+8..k0+15} U {k0+24..k0+31}
__device__ __forceinline__ v16bf load_a_frag(const unsigned short* A, int lda,
                                             int row0, int k0, int lane) {
  int half = lane >> 4;
  int m    = lane & 15;
  const unsigned short* p = A + (size_t)(row0 + m) * lda + k0 + half * 8;
  union { v16bf v; unsigned u[8]; } r;
#pragma unroll
  for (int i = 0; i < 4; ++i) r.u[i]     = *(const unsigned*)(p + 2 * i);
#pragma unroll
  for (int i = 0; i < 4; ++i) r.u[4 + i] = *(const unsigned*)(p + 16 + 2 * i);
  return r.v;
}

// Same A fragment, but sourced from an LDS-staged tile (tile-local row0).
__device__ __forceinline__ v16bf load_a_frag_lds(const unsigned short* sA, int lda,
                                                 int k0, int lane) {
  int half = lane >> 4;
  int m    = lane & 15;
  const unsigned short* p = sA + m * lda + k0 + half * 8;
  union { v16bf v; unsigned u[8]; } r;
#pragma unroll
  for (int i = 0; i < 4; ++i) r.u[i]     = *(const unsigned*)(p + 2 * i);
#pragma unroll
  for (int i = 0; i < 4; ++i) r.u[4 + i] = *(const unsigned*)(p + 16 + 2 * i);
  return r.v;
}

// B-matrix fragment (32x16 bf16) built from W^T stored row-major (N x K):
// lane<16: col n=lane, K = {k0..k0+7} U {k0+16..k0+23}; lane>=16: col n=lane-16, K = {k0+8..k0+15} U {k0+24..k0+31}
__device__ __forceinline__ v16bf load_b_frag(const unsigned short* WT, int ldb,
                                             int n0, int k0, int lane) {
  int half = lane >> 4;
  int n    = lane & 15;
  const unsigned short* p = WT + (size_t)(n0 + n) * ldb + k0 + half * 8;
  union { v16bf v; unsigned u[8]; } r;
#pragma unroll
  for (int i = 0; i < 8; ++i) r.u[i] = *(const unsigned*)(p + 2 * i + (i >> 2) * 8);
  return r.v;
}

__device__ __forceinline__ v8f wmma_bf16(v16bf a, v16bf b, v8f c) {
  // D = A x B + C, fp32 accumulate; emits v_wmma_f32_16x16x32_bf16
  return __builtin_amdgcn_wmma_f32_16x16x32_bf16(false, a, false, b,
                                                 (short)0, c, false, false);
}

__device__ __forceinline__ float sigmoidf(float x) {
  return 1.0f / (1.0f + __expf(-x));
}

// ---------------------------------------------------------------------------
// Tensor Data Mover: 1-D contiguous copy global -> LDS.
// Builds a minimal D# (CDNA5 ISA): data_size = 8 bytes,
// 1-D tile of `nbytes/8` units (<= 65535), tensor == tile so no OOB clipping.
// Issued once per wave (EXEC ignored); caller gates to one wave and then
// waits on TENSORcnt before the workgroup barrier.
// This toolchain exposes the 6-arg builtin:
//   (uint32x4 g0, int32x8 g1, int32x4 g2, int32x4 g3, int32x8 g4, i32 cpol)
// ---------------------------------------------------------------------------
__device__ __forceinline__ void tdm_copy_1d(const void* gptr, unsigned lds_off,
                                            unsigned nbytes) {
  unsigned long long ga = (unsigned long long)(uintptr_t)gptr;
  unsigned units = nbytes >> 3;                 // 8-byte units, must fit 16 bits
  v4u g0;
  g0.x = 1u;                                    // count=1, user descriptor
  g0.y = lds_off;                               // D#.lds_addr (bytes)
  g0.z = (unsigned)(ga & 0xFFFFFFFFu);          // global_addr[31:0]
  g0.w = (unsigned)((ga >> 32) & 0x01FFFFFFu)   // global_addr[56:32]
         | 0x80000000u;                         // type = 2 ("image")
  v8i g1;
  g1[0] = (int)(3u << 16);                      // wg_mask=0, data_size=3 (8B)
  g1[1] = (int)((units & 0xFFFFu) << 16);       // tensor_dim0[15:0]
  g1[2] = (int)((units >> 16) | (1u << 16));    // tensor_dim0[31:16], tensor_dim1=1
  g1[3] = (int)((units & 0xFFFFu) << 16);       // tensor_dim1[31:16]=0, tile_dim0=units
  g1[4] = 0;                                    // tile_dim1=0 (unused), tile_dim2=0
  g1[5] = (int)units;                           // tensor_dim0_stride[31:0]
  g1[6] = (int)((units & 0xFFFFu) << 16);       // stride0[47:32]=0, stride1[15:0]
  g1[7] = (int)(units >> 16);                   // tensor_dim1_stride[47:16]
  v4i z4 = {0, 0, 0, 0};
  v8i z8 = {0, 0, 0, 0, 0, 0, 0, 0};
  __builtin_amdgcn_tensor_load_to_lds(g0, g1, z4, z4, z8, 0);
}

__device__ __forceinline__ unsigned lds_addr32(const void* p) {
  // LDS aperture: flat address [31:0] is the workgroup-relative LDS offset.
  return (unsigned)(uintptr_t)p;
}

// ---------------------------------------------------------------------------
// prep kernels: one-time bf16 conversion / transposition of weights + inputs
// ---------------------------------------------------------------------------
__global__ void k_f2bf(const float* __restrict__ src, unsigned short* __restrict__ dst, int n) {
  int i = blockIdx.x * blockDim.x + threadIdx.x;
  if (i < n) dst[i] = f2bf_us(src[i]);
}

__global__ void k_copy_f32(const float* __restrict__ src, float* __restrict__ dst, int n) {
  int i = blockIdx.x * blockDim.x + threadIdx.x;
  if (i < n) dst[i] = src[i];
}

// src: K x N row-major fp32 ->  dst: N x K row-major bf16 (W^T)
__global__ void k_transpose_f2bf(const float* __restrict__ src, unsigned short* __restrict__ dst,
                                 int K, int N) {
  int i = blockIdx.x * blockDim.x + threadIdx.x;
  int total = K * N;
  if (i < total) {
    int k = i / N, n = i - k * N;
    dst[(size_t)n * K + k] = f2bf_us(src[i]);
  }
}

// embed_bf[b,t,h] = bf16(emb[inputs[b,t], h])
__global__ void k_embed(const int* __restrict__ tok, const float* __restrict__ emb,
                        unsigned short* __restrict__ dst) {
  int i = blockIdx.x * blockDim.x + threadIdx.x;   // over B*T*H
  if (i < B_ * T_ * H_) {
    int bt = i / H_, h = i - bt * H_;
    int v = tok[bt];
    dst[i] = f2bf_us(emb[(size_t)v * H_ + h]);
  }
}

// ---------------------------------------------------------------------------
// kproj / vproj: (B*S, H) = ann(B*S,H) @ Wk/Wv(H,H) + bias   (WMMA)
// grid (M/16, (H/16)/4), block 256 (8 waves): A tile (16x512 bf16 = 16 KB,
// contiguous) staged in LDS by the TDM, shared by all 8 waves.
// waves 0-3 -> 4 kproj n-tiles, waves 4-7 -> 4 vproj n-tiles.
// ---------------------------------------------------------------------------
__global__ void kv_proj_kernel(const unsigned short* __restrict__ ann_bf,
                               const unsigned short* __restrict__ WkT,
                               const float* __restrict__ bk,
                               const unsigned short* __restrict__ WvT,
                               const float* __restrict__ bv,
                               float* __restrict__ kproj, float* __restrict__ vproj) {
  __shared__ __attribute__((aligned(16))) unsigned short sA[16 * H_];
  int lane = threadIdx.x & 31;
  int wave = threadIdx.x >> 5;
  int m0   = blockIdx.x * 16;

  if (wave == 0) {
    tdm_copy_1d(ann_bf + (size_t)m0 * H_, lds_addr32(sA), 16 * H_ * 2);
    __builtin_amdgcn_s_wait_tensorcnt(0);
  }
  __syncthreads();

  int mat = wave >> 2;                        // 0 = kproj, 1 = vproj
  int n0  = (blockIdx.y * 4 + (wave & 3)) * 16;
  const unsigned short* WT = mat ? WvT : WkT;
  const float* bias        = mat ? bv  : bk;
  float* out               = mat ? vproj : kproj;

  v8f acc = {};
#pragma unroll 4
  for (int k = 0; k < H_; k += 32)
    acc = wmma_bf16(load_a_frag_lds(sA, H_, k, lane),
                    load_b_frag(WT, H_, n0, k, lane), acc);
  int n = n0 + (lane & 15);
  float bb = bias[n];
#pragma unroll
  for (int i = 0; i < 8; ++i) {
    int m = m0 + (lane >> 4) * 8 + i;
    out[(size_t)m * H_ + n] = acc[i] + bb;
  }
}

// ---------------------------------------------------------------------------
// per-step attention: q = h@Wq+bq ; scores = q.k * 1/sqrt(H); softmax; ctx;
// writes attn to output tail and xc (bf16 concat [embed_t, ctx]).
// grid = B blocks of 256 threads. Small (~21 MFLOP) latency-bound VALU work.
// ---------------------------------------------------------------------------
__global__ void attn_step_kernel(const float* __restrict__ h_cur,
                                 const float* __restrict__ Wq, const float* __restrict__ bq,
                                 const float* __restrict__ kproj, const float* __restrict__ vproj,
                                 const unsigned short* __restrict__ embed_bf,
                                 unsigned short* __restrict__ xc_bf,
                                 float* __restrict__ attn_out, int t) {
  __shared__ float h_s[H_];
  __shared__ float q_s[H_];
  __shared__ float sc4[S_ * 4];
  __shared__ float sc[S_];
  __shared__ float red[2];

  int b = blockIdx.x, tid = threadIdx.x;

  for (int i = tid; i < H_; i += 256) h_s[i] = h_cur[b * H_ + i];
  __syncthreads();

  // q = h @ Wq + bq   (Wq fp32, coalesced across n)
  for (int n = tid; n < H_; n += 256) {
    float acc = bq[n];
    for (int k = 0; k < H_; ++k) acc = __fmaf_rn(h_s[k], Wq[k * H_ + n], acc);
    q_s[n] = acc;
  }
  __syncthreads();

  // scores: 4 threads per source position
  {
    int s = tid >> 2, part = tid & 3;
    const float* kp = kproj + ((size_t)(b * S_ + s)) * H_ + part * 128;
    const float* qp = q_s + part * 128;
    float acc = 0.f;
    for (int k = 0; k < 128; ++k) acc = __fmaf_rn(qp[k], kp[k], acc);
    sc4[tid] = acc;
  }
  __syncthreads();
  if (tid < S_)
    sc[tid] = (sc4[tid * 4] + sc4[tid * 4 + 1] + sc4[tid * 4 + 2] + sc4[tid * 4 + 3])
              * 0.04419417382415922f;  // 1/sqrt(512)
  __syncthreads();
  if (tid == 0) {
    float m = sc[0];
    for (int s = 1; s < S_; ++s) m = fmaxf(m, sc[s]);
    red[0] = m;
  }
  __syncthreads();
  if (tid < S_) sc[tid] = __expf(sc[tid] - red[0]);
  __syncthreads();
  if (tid == 0) {
    float s0 = 0.f;
    for (int s = 0; s < S_; ++s) s0 += sc[s];
    red[1] = 1.f / s0;
  }
  __syncthreads();
  if (tid < S_) {
    float a = sc[tid] * red[1];
    sc[tid] = a;
    attn_out[(size_t)b * (T_ * S_) + t * S_ + tid] = a;
  }
  __syncthreads();

  // ctx = attn @ vproj ; xc = [embed_t, ctx] as bf16
  for (int n = tid; n < H_; n += 256) {
    float acc = 0.f;
    for (int s = 0; s < S_; ++s)
      acc = __fmaf_rn(sc[s], vproj[((size_t)(b * S_ + s)) * H_ + n], acc);
    xc_bf[(size_t)b * (2 * H_) + H_ + n] = f2bf_us(acc);
    xc_bf[(size_t)b * (2 * H_) + n]      = embed_bf[((size_t)(b * T_ + t)) * H_ + n];
  }
}

// ---------------------------------------------------------------------------
// per-step GRU: all 6 gate GEMMs via WMMA, fused nonlinearity + state update.
// grid = H/16 = 32 blocks (N tiles), block 64 = 2 waves (M rows 0-15 / 16-31).
// Each A fragment is consumed by 3 WMMAs (one per gate) from registers, so
// LDS staging buys nothing here; weights stream from L2 (4.5 MB/step).
// ---------------------------------------------------------------------------
__global__ void gru_step_kernel(const unsigned short* __restrict__ xc_bf,
                                const unsigned short* __restrict__ h_bf,
                                const float* __restrict__ h_cur,
                                const unsigned short* __restrict__ WxrT,
                                const unsigned short* __restrict__ WhrT, const float* __restrict__ bhr,
                                const unsigned short* __restrict__ WxzT,
                                const unsigned short* __restrict__ WhzT, const float* __restrict__ bhz,
                                const unsigned short* __restrict__ WxgT,
                                const unsigned short* __restrict__ WhgT, const float* __restrict__ bhg,
                                float* __restrict__ h_next,
                                unsigned short* __restrict__ h_next_bf,
                                unsigned short* __restrict__ hiddens_bf, int t) {
  int lane = threadIdx.x & 31;
  int m0   = (threadIdx.x >> 5) * 16;
  int n0   = blockIdx.x * 16;

  v8f ar = {}, az = {}, ag = {}, hr = {}, hz = {}, hg = {};

#pragma unroll 2
  for (int k = 0; k < 2 * H_; k += 32) {
    v16bf a = load_a_frag(xc_bf, 2 * H_, m0, k, lane);
    ar = wmma_bf16(a, load_b_frag(WxrT, 2 * H_, n0, k, lane), ar);
    az = wmma_bf16(a, load_b_frag(WxzT, 2 * H_, n0, k, lane), az);
    ag = wmma_bf16(a, load_b_frag(WxgT, 2 * H_, n0, k, lane), ag);
  }
#pragma unroll 2
  for (int k = 0; k < H_; k += 32) {
    v16bf a = load_a_frag(h_bf, H_, m0, k, lane);
    hr = wmma_bf16(a, load_b_frag(WhrT, H_, n0, k, lane), hr);
    hz = wmma_bf16(a, load_b_frag(WhzT, H_, n0, k, lane), hz);
    hg = wmma_bf16(a, load_b_frag(WhgT, H_, n0, k, lane), hg);
  }

  int n  = n0 + (lane & 15);
  int mh = (lane >> 4) * 8;
  float vbhr = bhr[n], vbhz = bhz[n], vbhg = bhg[n];
#pragma unroll
  for (int i = 0; i < 8; ++i) {
    int m   = m0 + mh + i;          // batch index
    float r = sigmoidf(ar[i] + hr[i] + vbhr);
    float z = sigmoidf(az[i] + hz[i] + vbhz);
    float g = tanhf(ag[i] + r * (hg[i] + vbhg));
    float hold = h_cur[(size_t)m * H_ + n];
    float hn   = (1.0f - z) * g + z * hold;
    h_next[(size_t)m * H_ + n]                 = hn;
    unsigned short hb                          = f2bf_us(hn);
    h_next_bf[(size_t)m * H_ + n]              = hb;
    hiddens_bf[((size_t)m * T_ + t) * H_ + n]  = hb;
  }
}

// ---------------------------------------------------------------------------
// output projection: (B*T, V) = hiddens(B*T,H) @ Wout(H,V) + bout  (67 GFLOP)
// grid (M/64, V/128), block 256 (8 waves): a 64x512 bf16 A panel (64 KB,
// contiguous) is staged in LDS by one TDM op and reused by all 8 waves.
// Each wave owns one 16-column strip and 4 M sub-tiles, so every B fragment
// (64 B from L2) feeds 4 WMMAs — 4x less WoutT L2 traffic than M=16 blocking
// (~1 GB total), pushing the kernel firmly into the compute-bound regime.
// ---------------------------------------------------------------------------
__global__ void out_proj_kernel(const unsigned short* __restrict__ hid_bf,
                                const unsigned short* __restrict__ WoutT,
                                const float* __restrict__ bout,
                                float* __restrict__ out) {
  __shared__ __attribute__((aligned(16))) unsigned short sA[64 * H_];  // 64 KB
  int lane = threadIdx.x & 31;
  int wave = threadIdx.x >> 5;
  int m0   = blockIdx.x * 64;

  if (wave == 0) {
    tdm_copy_1d(hid_bf + (size_t)m0 * H_, lds_addr32(sA), 64 * H_ * 2);
    __builtin_amdgcn_s_wait_tensorcnt(0);
  }
  __syncthreads();

  int n0 = (blockIdx.y * 8 + wave) * 16;
  v8f acc0 = {}, acc1 = {}, acc2 = {}, acc3 = {};
#pragma unroll 2
  for (int k = 0; k < H_; k += 32) {
    v16bf bfr = load_b_frag(WoutT, H_, n0, k, lane);
    acc0 = wmma_bf16(load_a_frag_lds(sA +  0 * H_, H_, k, lane), bfr, acc0);
    acc1 = wmma_bf16(load_a_frag_lds(sA + 16 * H_, H_, k, lane), bfr, acc1);
    acc2 = wmma_bf16(load_a_frag_lds(sA + 32 * H_, H_, k, lane), bfr, acc2);
    acc3 = wmma_bf16(load_a_frag_lds(sA + 48 * H_, H_, k, lane), bfr, acc3);
  }

  int n  = n0 + (lane & 15);
  int mh = (lane >> 4) * 8;
  float bb = bout[n];
#pragma unroll
  for (int i = 0; i < 8; ++i) {
    int m = m0 + mh + i;
    out[(size_t)(m +  0) * V_ + n] = acc0[i] + bb;
    out[(size_t)(m + 16) * V_ + n] = acc1[i] + bb;
    out[(size_t)(m + 32) * V_ + n] = acc2[i] + bb;
    out[(size_t)(m + 48) * V_ + n] = acc3[i] + bb;
  }
}

// ---------------------------------------------------------------------------
// orchestration
// ---------------------------------------------------------------------------
extern "C" void kernel_launch(void* const* d_in, const int* in_sizes, int n_in,
                              void* d_out, int out_size, void* d_ws, size_t ws_size,
                              hipStream_t stream) {
  (void)in_sizes; (void)n_in; (void)out_size; (void)ws_size;

  const int*   inputs      = (const int*)d_in[0];
  const float* annotations = (const float*)d_in[1];
  const float* hidden_init = (const float*)d_in[2];
  const float* emb         = (const float*)d_in[3];
  const float* Wq  = (const float*)d_in[4];
  const float* bq  = (const float*)d_in[5];
  const float* Wk  = (const float*)d_in[6];
  const float* bk  = (const float*)d_in[7];
  const float* Wv  = (const float*)d_in[8];
  const float* bv  = (const float*)d_in[9];
  const float* Wxr = (const float*)d_in[10];
  const float* Whr = (const float*)d_in[11];
  const float* bhr = (const float*)d_in[12];
  const float* Wxz = (const float*)d_in[13];
  const float* Whz = (const float*)d_in[14];
  const float* bhz = (const float*)d_in[15];
  const float* Wxg = (const float*)d_in[16];
  const float* Whg = (const float*)d_in[17];
  const float* bhg = (const float*)d_in[18];
  const float* Wout = (const float*)d_in[19];
  const float* bout = (const float*)d_in[20];

  // ---- workspace layout (bytes) --------------------------------------------
  char* ws = (char*)d_ws;
  size_t off = 0;
  auto take = [&](size_t bytes) { char* p = ws + off; off += (bytes + 255) & ~(size_t)255; return p; };

  float* kproj = (float*)take(sizeof(float) * B_ * S_ * H_);
  float* vproj = (float*)take(sizeof(float) * B_ * S_ * H_);
  float* h_f32 = (float*)take(sizeof(float) * 2 * B_ * H_);           // ping-pong

  unsigned short* embed_bf   = (unsigned short*)take(2ull * B_ * T_ * H_);
  unsigned short* ann_bf     = (unsigned short*)take(2ull * B_ * S_ * H_);
  unsigned short* hiddens_bf = (unsigned short*)take(2ull * B_ * T_ * H_);
  unsigned short* h_bf       = (unsigned short*)take(2ull * 2 * B_ * H_); // ping-pong
  unsigned short* xc_bf      = (unsigned short*)take(2ull * B_ * 2 * H_);

  unsigned short* WkT  = (unsigned short*)take(2ull * H_ * H_);
  unsigned short* WvT  = (unsigned short*)take(2ull * H_ * H_);
  unsigned short* WxrT = (unsigned short*)take(2ull * 2 * H_ * H_);
  unsigned short* WxzT = (unsigned short*)take(2ull * 2 * H_ * H_);
  unsigned short* WxgT = (unsigned short*)take(2ull * 2 * H_ * H_);
  unsigned short* WhrT = (unsigned short*)take(2ull * H_ * H_);
  unsigned short* WhzT = (unsigned short*)take(2ull * H_ * H_);
  unsigned short* WhgT = (unsigned short*)take(2ull * H_ * H_);
  unsigned short* WoutT = (unsigned short*)take(2ull * (size_t)V_ * H_);

  float* out_logits = (float*)d_out;                              // (B,T,V)
  float* attn_out   = (float*)d_out + (size_t)B_ * T_ * V_;       // (B, T*S)

  const int TPB = 256;
  auto blocks = [](int n, int tpb) { return (n + tpb - 1) / tpb; };

  // ---- prep: bf16 conversions + weight transposes --------------------------
  k_transpose_f2bf<<<blocks(H_ * H_, TPB), TPB, 0, stream>>>(Wk,  WkT,  H_, H_);
  k_transpose_f2bf<<<blocks(H_ * H_, TPB), TPB, 0, stream>>>(Wv,  WvT,  H_, H_);
  k_transpose_f2bf<<<blocks(2 * H_ * H_, TPB), TPB, 0, stream>>>(Wxr, WxrT, 2 * H_, H_);
  k_transpose_f2bf<<<blocks(2 * H_ * H_, TPB), TPB, 0, stream>>>(Wxz, WxzT, 2 * H_, H_);
  k_transpose_f2bf<<<blocks(2 * H_ * H_, TPB), TPB, 0, stream>>>(Wxg, WxgT, 2 * H_, H_);
  k_transpose_f2bf<<<blocks(H_ * H_, TPB), TPB, 0, stream>>>(Whr, WhrT, H_, H_);
  k_transpose_f2bf<<<blocks(H_ * H_, TPB), TPB, 0, stream>>>(Whz, WhzT, H_, H_);
  k_transpose_f2bf<<<blocks(H_ * H_, TPB), TPB, 0, stream>>>(Whg, WhgT, H_, H_);
  k_transpose_f2bf<<<blocks(H_ * V_, TPB), TPB, 0, stream>>>(Wout, WoutT, H_, V_);

  k_f2bf<<<blocks(B_ * S_ * H_, TPB), TPB, 0, stream>>>(annotations, ann_bf, B_ * S_ * H_);
  k_f2bf<<<blocks(B_ * H_, TPB), TPB, 0, stream>>>(hidden_init, h_bf, B_ * H_);
  k_copy_f32<<<blocks(B_ * H_, TPB), TPB, 0, stream>>>(hidden_init, h_f32, B_ * H_);
  k_embed<<<blocks(B_ * T_ * H_, TPB), TPB, 0, stream>>>(inputs, emb, embed_bf);

  // ---- K/V projections (step-invariant) ------------------------------------
  kv_proj_kernel<<<dim3((B_ * S_) / 16, (H_ / 16) / 4), 256, 0, stream>>>(
      ann_bf, WkT, bk, WvT, bv, kproj, vproj);

  // ---- sequential recurrence over T ----------------------------------------
  for (int t = 0; t < T_; ++t) {
    const float*          hc  = h_f32 + (t & 1) * B_ * H_;
    const unsigned short* hcb = h_bf  + (t & 1) * B_ * H_;
    float*                hn  = h_f32 + ((t + 1) & 1) * B_ * H_;
    unsigned short*       hnb = h_bf  + ((t + 1) & 1) * B_ * H_;

    attn_step_kernel<<<B_, 256, 0, stream>>>(hc, Wq, bq, kproj, vproj,
                                             embed_bf, xc_bf, attn_out, t);
    gru_step_kernel<<<H_ / 16, 64, 0, stream>>>(xc_bf, hcb, hc,
                                                WxrT, WhrT, bhr,
                                                WxzT, WhzT, bhz,
                                                WxgT, WhgT, bhg,
                                                hn, hnb, hiddens_bf, t);
  }

  // ---- output projection (dominant GEMM, 67 GFLOP) -------------------------
  out_proj_kernel<<<dim3((B_ * T_) / 64, V_ / 128), 256, 0, stream>>>(
      hiddens_bf, WoutT, bout, out_logits);
}
